// AgentAttention_79740362817548
// MI455X (gfx1250) — compile-verified, matrix operand order checked by reading, not run on hardware
//
#include <hip/hip_runtime.h>
#include <hip/hip_bf16.h>

// ---------------------------------------------------------------------------
// Agent Attention (B=1024, N=A=169, C=256, H=4, HD=64) for gfx1250 / MI455X
// bf16 WMMA (v_wmma_f32_16x16x32_bf16), fp32 softmax/accum, wave32, 11 waves/WG
// Async global->LDS staging (global_load_async_to_lds_b128 / s_wait_asynccnt)
// ---------------------------------------------------------------------------

typedef __attribute__((ext_vector_type(16))) __bf16 v16bf;
typedef __attribute__((ext_vector_type(8)))  __bf16 v8bf;
typedef __attribute__((ext_vector_type(4)))  __bf16 v4bf;
typedef __attribute__((ext_vector_type(8)))  float  v8f;
typedef __attribute__((ext_vector_type(4)))  float  v4f;

#define WMMA_BF16(A_, B_, C_) \
  __builtin_amdgcn_wmma_f32_16x16x32_bf16(false, (A_), false, (B_), (short)0, (C_), false, false)

#define V8F_ZERO {0.f,0.f,0.f,0.f,0.f,0.f,0.f,0.f}

// Shapes / constants
#define BATCH 1024
#define NTOK  169      // N == A
#define CDIM  256
#define HEADS 4
#define HD    64
#define WIMG  13
#define NPAD  176      // 11 * 16
#define LDX   264      // x tile leading dim (bf16, 16B-aligned rows)
#define LDQ   72       // q/k leading dim
#define LDP   192      // probs / transposed v / transposed agent_v leading dim

// LDS layout (bytes)
#define OFF_XS   0                           // [176][264] bf16 (92928 B) -- reused as Ps [176][192]
#define OFF_QS   92928                       // [176][72]  bf16 (25344 B)
#define OFF_KS   118272                      // [176][72]  bf16 (25344 B)
#define OFF_VST  143616                      // [72][192]  bf16 (27648 B)  v transposed (d-major)
#define OFF_AVT  171264                      // [72][192]  bf16 (27648 B)  agent_v transposed
#define OFF_WSL  198912                      // [192][256] bf16 (98304 B)  head qkv weight slice
#define LDS_ATTN 297216
#define LDS_PROJ 92928                       // [176][264] bf16

// Workspace layout (bytes)
#define WS_QWB   0                           // qkv_w bf16 : 768*256*2  = 393216
#define WS_PWB   393216                      // proj_w bf16: 256*256*2  = 131072
#define WS_POSB  524288                      // pos bias   : 4*176*176*4 = 495616
#define WS_AGB   1019904                     // agent bias : 4*176*176*4 = 495616
#define WS_Y     1515520                     // y bf16     : 1024*169*256*2 = 88604672

// ---------------------------------------------------------------------------
// Async 16-byte global -> LDS copy (ASYNCcnt-tracked, no VGPR data movement).
// lds_off: LDS byte offset (low 32 bits of a generic shared pointer).
// ---------------------------------------------------------------------------
__device__ __forceinline__ void async_copy_b128(unsigned lds_off, const void* gsrc) {
  asm volatile("global_load_async_to_lds_b128 %0, %1, off"
               :: "v"(lds_off), "v"(gsrc)
               : "memory");
}
__device__ __forceinline__ void wait_async0() {
  asm volatile("s_wait_asynccnt 0" ::: "memory");
}

// ---------------------------------------------------------------------------
// Fragment loader.  Per CDNA5 ISA 7.12.2 (16-bit, 16x32):
//   row/col = lane&15 ; K indices = (lane>>4)*8 + {0..7} and + {16..23}
// Source array must be "operand-major" with K contiguous (ld = K-stride).
// Two contiguous 16B loads per lane.
// ---------------------------------------------------------------------------
__device__ __forceinline__ v16bf ldfrag(const __bf16* p, int ld, int lane) {
  const __bf16* q = p + (lane & 15) * ld + ((lane >> 4) << 3);
  union { v8bf h[2]; v16bf v; } u;
  u.h[0] = *(const v8bf*)q;
  u.h[1] = *(const v8bf*)(q + 16);
  return u.v;
}

// Row softmax over an 11-tile strip of scores held in D-layout accumulators,
// then store probabilities (bf16) to Ps[m][n] (ld = LDP), zeroing pad columns.
__device__ __forceinline__ void softmax_store(v8f* acc, const float* __restrict__ bias,
                                              __bf16* Ps, int m0, int lane, float scale) {
  const int hi  = lane >> 4;
  const int col = lane & 15;
  float rmax[8], rsum[8];
#pragma unroll
  for (int r = 0; r < 8; ++r) rmax[r] = -3.0e38f;
#pragma unroll
  for (int nt = 0; nt < 11; ++nt) {
    const int n = (nt << 4) + col;
#pragma unroll
    for (int r = 0; r < 8; ++r) {
      const int m = m0 + r + (hi << 3);
      float s = acc[nt][r] * scale + bias[m * NPAD + n];
      s = (n < NTOK) ? s : -3.0e38f;
      acc[nt][r] = s;
      rmax[r] = fmaxf(rmax[r], s);
    }
  }
#pragma unroll
  for (int r = 0; r < 8; ++r) {
#pragma unroll
    for (int mk = 1; mk < 16; mk <<= 1)
      rmax[r] = fmaxf(rmax[r], __shfl_xor(rmax[r], mk, 32));
    rsum[r] = 0.f;
  }
#pragma unroll
  for (int nt = 0; nt < 11; ++nt) {
#pragma unroll
    for (int r = 0; r < 8; ++r) {
      const float e = __expf(acc[nt][r] - rmax[r]);
      acc[nt][r] = e;
      rsum[r] += e;
    }
  }
#pragma unroll
  for (int r = 0; r < 8; ++r) {
#pragma unroll
    for (int mk = 1; mk < 16; mk <<= 1)
      rsum[r] += __shfl_xor(rsum[r], mk, 32);
    rsum[r] = 1.f / rsum[r];
  }
#pragma unroll
  for (int nt = 0; nt < 11; ++nt) {
    const int n = (nt << 4) + col;
#pragma unroll
    for (int r = 0; r < 8; ++r) {
      const int m = m0 + r + (hi << 3);
      Ps[m * LDP + n] = (__bf16)(acc[nt][r] * rsum[r]);
    }
  }
#pragma unroll
  for (int r = 0; r < 8; ++r)
    Ps[(m0 + r + (hi << 3)) * LDP + NPAD + col] = (__bf16)0.f;
}

// ---------------------------------------------------------------------------
// Pre-pass 1: fp32 -> bf16 weight conversion (qkv_w 768x256, proj_w 256x256)
// ---------------------------------------------------------------------------
__global__ void cvt_w_kernel(const float* __restrict__ qw, const float* __restrict__ pw,
                             __bf16* __restrict__ qwb, __bf16* __restrict__ pwb) {
  const int i = blockIdx.x * 256 + threadIdx.x;
  if (i < 768 * 256) qwb[i] = (__bf16)qw[i];
  if (i < 256 * 256) pwb[i] = (__bf16)pw[i];
}

// ---------------------------------------------------------------------------
// Pre-pass 2: fold the six bias tensors into padded [H][176][176] tables.
//   pos[h][a][n]  = an[h,a,n] + ah[h,a,n/13] + aw[h,a,n%13]
//   agb[h][n][a]  = na[h,a,n] + ha[h,n/13,a] + wa[h,n%13,a]
// ---------------------------------------------------------------------------
__global__ void bias_pre_kernel(const float* __restrict__ an, const float* __restrict__ na,
                                const float* __restrict__ ah, const float* __restrict__ aw,
                                const float* __restrict__ ha, const float* __restrict__ wa,
                                float* __restrict__ pos, float* __restrict__ agb) {
  const int idx = blockIdx.x * 256 + threadIdx.x;
  if (idx >= HEADS * NPAD * NPAD) return;
  const int h   = idx / (NPAD * NPAD);
  const int rem = idx - h * (NPAD * NPAD);
  const int r   = rem / NPAD;
  const int c   = rem - r * NPAD;
  float pv = 0.f, av = 0.f;
  if (r < NTOK && c < NTOK) {
    // pos: row = agent a, col = token n
    pv = an[h * (NTOK * NTOK) + r * NTOK + c]
       + ah[(h * NTOK + r) * WIMG + c / WIMG]
       + aw[(h * NTOK + r) * WIMG + c % WIMG];
    // agb: row = token n, col = agent a  (na transposed)
    av = na[h * (NTOK * NTOK) + c * NTOK + r]
       + ha[(h * WIMG + r / WIMG) * NTOK + c]
       + wa[(h * WIMG + r % WIMG) * NTOK + c];
  }
  pos[idx] = pv;
  agb[idx] = av;
}

// ---------------------------------------------------------------------------
// Main fused kernel: one workgroup per (b, h).  11 waves, wave w owns rows
// 16w..16w+15 of every 169-row operand.
// ---------------------------------------------------------------------------
__global__ __launch_bounds__(352) void attn_kernel(
    const float* __restrict__ x, const __bf16* __restrict__ qwb,
    const float* __restrict__ posb, const float* __restrict__ agb,
    const float* __restrict__ dwcw, const float* __restrict__ dwcb,
    __bf16* __restrict__ y) {
  extern __shared__ char smem[];
  __bf16* xs  = (__bf16*)(smem + OFF_XS);   // [176][264]
  __bf16* Ps  = (__bf16*)(smem + OFF_XS);   // [176][192]  (reuses xs)
  __bf16* qs  = (__bf16*)(smem + OFF_QS);   // [176][72]   token-major
  __bf16* ks  = (__bf16*)(smem + OFF_KS);   // [176][72]   token-major
  __bf16* vsT = (__bf16*)(smem + OFF_VST);  // [72][192]   d-major
  __bf16* avT = (__bf16*)(smem + OFF_AVT);  // [72][192]   d-major
  __bf16* wsl = (__bf16*)(smem + OFF_WSL);  // [192][256]  qkv weight slice for head h

  const int tid  = threadIdx.x;
  const int lane = tid & 31;
  const int wv   = tid >> 5;                // 0..10
  const int hi   = lane >> 4;
  const int col  = lane & 15;
  const int b    = blockIdx.x >> 2;
  const int h    = blockIdx.x & 3;
  const int m0   = wv << 4;
  const float scale = 0.125f;               // HD^-0.5

  // ---- kick off async staging of the head's qkv weight slice (raw bf16) ----
  // Three 64-row chunks (q,k,v) of 32768 B each; rows are 512 B.
  {
    const unsigned wsl_base = (unsigned)(size_t)(const void*)wsl;
    const char* wsrc = (const char*)qwb + (size_t)h * 32768;
    for (int idx = tid; idx < 6144; idx += 352) {
      const int chunk = idx >> 11;             // sel = 0..2
      const int off   = (idx & 2047) << 4;     // byte offset within chunk
      async_copy_b128(wsl_base + (unsigned)(chunk * 32768 + off),
                      wsrc + (size_t)chunk * 131072 + off);
    }
  }

  // ---- stage in: x(b) -> LDS bf16 (overlaps the async copy), zero pads ----
  const float* xb = x + (size_t)b * (NTOK * CDIM);
  for (int idx = tid; idx < NTOK * 64; idx += 352) {
    const int row = idx >> 6;
    const int c4  = (idx & 63) << 2;
    v4f f = *(const v4f*)(xb + row * CDIM + c4);
    v4bf o;
    o[0] = (__bf16)f[0]; o[1] = (__bf16)f[1];
    o[2] = (__bf16)f[2]; o[3] = (__bf16)f[3];
    *(v4bf*)(xs + row * LDX + c4) = o;
  }
  for (int idx = tid; idx < 7 * LDX; idx += 352) xs[NTOK * LDX + idx] = (__bf16)0.f;
  for (int idx = tid; idx < 72 * 16; idx += 352) {
    const int r = idx >> 4, cc = NPAD + (idx & 15);
    vsT[r * LDP + cc] = (__bf16)0.f;
    avT[r * LDP + cc] = (__bf16)0.f;
  }
  wait_async0();
  __syncthreads();

  // ---- stage 0: qkv projection for head h (q,k token-major; v d-major) ----
  {
    v16bf a8[8];
#pragma unroll
    for (int k8 = 0; k8 < 8; ++k8)
      a8[k8] = ldfrag(xs + m0 * LDX + (k8 << 5), LDX, lane);
#pragma unroll
    for (int sel = 0; sel < 3; ++sel) {
#pragma unroll
      for (int nt = 0; nt < 4; ++nt) {
        v8f acc = V8F_ZERO;
#pragma unroll
        for (int k8 = 0; k8 < 8; ++k8) {
          v16bf bb = ldfrag(wsl + (sel * 64 + nt * 16) * CDIM + (k8 << 5), CDIM, lane);
          acc = WMMA_BF16(a8[k8], bb, acc);
        }
#pragma unroll
        for (int r = 0; r < 8; ++r) {
          const int m = m0 + r + (hi << 3);
          const int d = (nt << 4) + col;
          const __bf16 v = (__bf16)acc[r];
          if (sel == 0)      qs[m * LDQ + d] = v;
          else if (sel == 1) ks[m * LDQ + d] = v;
          else               vsT[d * LDP + m] = v;
        }
      }
    }
  }
  __syncthreads();

  // ---- GEMM1: S1 = scale * a(q) @ k^T + pos_bias ; softmax -> Ps ----
  {
    v8f acc[11];
#pragma unroll
    for (int nt = 0; nt < 11; ++nt) acc[nt] = (v8f)V8F_ZERO;
#pragma unroll
    for (int k2 = 0; k2 < 2; ++k2) {
      const int kk = k2 << 5;
      v16bf a = ldfrag(qs + m0 * LDQ + kk, LDQ, lane);
#pragma unroll
      for (int nt = 0; nt < 11; ++nt) {
        v16bf bb = ldfrag(ks + (nt * 16) * LDQ + kk, LDQ, lane);
        acc[nt] = WMMA_BF16(a, bb, acc[nt]);
      }
    }
    softmax_store(acc, posb + h * (NPAD * NPAD), Ps, m0, lane, scale);
  }
  __syncthreads();

  // ---- GEMM2: agent_v = P1 @ v  (K = tokens, padded to 192) -> avT ----
  {
    v8f acc4[4];
#pragma unroll
    for (int nt = 0; nt < 4; ++nt) acc4[nt] = (v8f)V8F_ZERO;
#pragma unroll
    for (int k6 = 0; k6 < 6; ++k6) {
      const int kk = k6 << 5;
      v16bf a = ldfrag(Ps + m0 * LDP + kk, LDP, lane);
#pragma unroll
      for (int nt = 0; nt < 4; ++nt) {
        v16bf bb = ldfrag(vsT + (nt * 16) * LDP + kk, LDP, lane);
        acc4[nt] = WMMA_BF16(a, bb, acc4[nt]);
      }
    }
#pragma unroll
    for (int nt = 0; nt < 4; ++nt) {
      const int d = (nt << 4) + col;
#pragma unroll
      for (int r = 0; r < 8; ++r)
        avT[d * LDP + (m0 + r + (hi << 3))] = (__bf16)acc4[nt][r];
    }
  }
  __syncthreads();

  // ---- GEMM3: S2 = scale * q @ a^T + agent_bias ; softmax -> Ps (reuse) ----
  {
    v8f acc[11];
#pragma unroll
    for (int nt = 0; nt < 11; ++nt) acc[nt] = (v8f)V8F_ZERO;
#pragma unroll
    for (int k2 = 0; k2 < 2; ++k2) {
      const int kk = k2 << 5;
      v16bf a = ldfrag(qs + m0 * LDQ + kk, LDQ, lane);
#pragma unroll
      for (int nt = 0; nt < 11; ++nt) {
        v16bf bb = ldfrag(qs + (nt * 16) * LDQ + kk, LDQ, lane);  // a == q
        acc[nt] = WMMA_BF16(a, bb, acc[nt]);
      }
    }
    softmax_store(acc, agb + h * (NPAD * NPAD), Ps, m0, lane, scale);
  }
  __syncthreads();

  // ---- GEMM4: out = P2 @ agent_v ; fuse 3x3 depthwise conv on v ; store y ----
  {
    v8f acc4[4];
#pragma unroll
    for (int nt = 0; nt < 4; ++nt) acc4[nt] = (v8f)V8F_ZERO;
#pragma unroll
    for (int k6 = 0; k6 < 6; ++k6) {
      const int kk = k6 << 5;
      v16bf a = ldfrag(Ps + m0 * LDP + kk, LDP, lane);
#pragma unroll
      for (int nt = 0; nt < 4; ++nt) {
        v16bf bb = ldfrag(avT + (nt * 16) * LDP + kk, LDP, lane);
        acc4[nt] = WMMA_BF16(a, bb, acc4[nt]);
      }
    }
    __bf16* yb = y + (size_t)b * (NTOK * CDIM);
#pragma unroll
    for (int nt = 0; nt < 4; ++nt) {
      const int d = (nt << 4) + col;
      const int c = h * HD + d;
      float w9[9];
#pragma unroll
      for (int t = 0; t < 9; ++t) w9[t] = dwcw[c * 9 + t];
      const float cb = dwcb[c];
#pragma unroll
      for (int r = 0; r < 8; ++r) {
        const int m = m0 + r + (hi << 3);
        if (m < NTOK) {
          const int i = m / WIMG;
          const int j = m - i * WIMG;
          float dv = cb;
#pragma unroll
          for (int di = 0; di < 3; ++di) {
            const int ii = i + di - 1;
            if (ii >= 0 && ii < WIMG) {
#pragma unroll
              for (int dj = 0; dj < 3; ++dj) {
                const int jj = j + dj - 1;
                if (jj >= 0 && jj < WIMG)
                  dv += w9[di * 3 + dj] * (float)vsT[d * LDP + ii * WIMG + jj];
              }
            }
          }
          yb[m * CDIM + c] = (__bf16)(acc4[nt][r] + dv);
        }
      }
    }
  }
}

// ---------------------------------------------------------------------------
// Final projection: out = y @ proj_w^T + proj_b   (one block per batch b)
// ---------------------------------------------------------------------------
__global__ __launch_bounds__(352) void proj_kernel(
    const __bf16* __restrict__ y, const __bf16* __restrict__ pwb,
    const float* __restrict__ pb, float* __restrict__ out) {
  extern __shared__ char smem[];
  __bf16* ys = (__bf16*)smem;               // [176][264]
  const int tid  = threadIdx.x;
  const int lane = tid & 31;
  const int wv   = tid >> 5;
  const int hi   = lane >> 4;
  const int col  = lane & 15;
  const int b    = blockIdx.x;
  const int m0   = wv << 4;

  // async-stage y(b) (bf16) into LDS with padded stride
  const __bf16* yb = y + (size_t)b * (NTOK * CDIM);
  {
    const unsigned ys_base = (unsigned)(size_t)(const void*)ys;
    for (int idx = tid; idx < NTOK * 32; idx += 352) {
      const int row = idx >> 5;
      const int c8  = (idx & 31) << 3;
      async_copy_b128(ys_base + (unsigned)((row * LDX + c8) * 2),
                      (const void*)(yb + row * CDIM + c8));
    }
  }
  for (int idx = tid; idx < 7 * LDX; idx += 352) ys[NTOK * LDX + idx] = (__bf16)0.f;
  wait_async0();
  __syncthreads();

  v16bf a8[8];
#pragma unroll
  for (int k8 = 0; k8 < 8; ++k8)
    a8[k8] = ldfrag(ys + m0 * LDX + (k8 << 5), LDX, lane);

  float* ob = out + (size_t)b * (NTOK * CDIM);
#pragma unroll
  for (int nt = 0; nt < 16; ++nt) {
    v8f acc = V8F_ZERO;
#pragma unroll
    for (int k8 = 0; k8 < 8; ++k8) {
      v16bf bb = ldfrag(pwb + (nt * 16) * CDIM + (k8 << 5), CDIM, lane);
      acc = WMMA_BF16(a8[k8], bb, acc);
    }
    const int n = (nt << 4) + col;
    const float bias = pb[n];
#pragma unroll
    for (int r = 0; r < 8; ++r) {
      const int m = m0 + r + (hi << 3);
      if (m < NTOK) ob[m * CDIM + n] = acc[r] + bias;
    }
  }
}

// ---------------------------------------------------------------------------
extern "C" void kernel_launch(void* const* d_in, const int* in_sizes, int n_in,
                              void* d_out, int out_size, void* d_ws, size_t ws_size,
                              hipStream_t stream) {
  const float* x      = (const float*)d_in[0];
  const float* qkv_w  = (const float*)d_in[1];
  const float* proj_w = (const float*)d_in[2];
  const float* proj_b = (const float*)d_in[3];
  const float* dwc_w  = (const float*)d_in[4];
  const float* dwc_b  = (const float*)d_in[5];
  const float* an_b   = (const float*)d_in[6];
  const float* na_b   = (const float*)d_in[7];
  const float* ah_b   = (const float*)d_in[8];
  const float* aw_b   = (const float*)d_in[9];
  const float* ha_b   = (const float*)d_in[10];
  const float* wa_b   = (const float*)d_in[11];
  float* outp = (float*)d_out;

  char* ws = (char*)d_ws;
  __bf16* qwb  = (__bf16*)(ws + WS_QWB);
  __bf16* pwb  = (__bf16*)(ws + WS_PWB);
  float*  posb = (float*)(ws + WS_POSB);
  float*  agb  = (float*)(ws + WS_AGB);
  __bf16* ybuf = (__bf16*)(ws + WS_Y);

  (void)in_sizes; (void)n_in; (void)out_size; (void)ws_size;

  // Allow > default dynamic LDS for the big fused kernel.
  hipFuncSetAttribute((const void*)attn_kernel,
                      hipFuncAttributeMaxDynamicSharedMemorySize, LDS_ATTN);
  hipFuncSetAttribute((const void*)proj_kernel,
                      hipFuncAttributeMaxDynamicSharedMemorySize, LDS_PROJ);

  cvt_w_kernel<<<768, 256, 0, stream>>>(qkv_w, proj_w, qwb, pwb);
  bias_pre_kernel<<<(HEADS * NPAD * NPAD + 255) / 256, 256, 0, stream>>>(
      an_b, na_b, ah_b, aw_b, ha_b, wa_b, posb, agb);
  attn_kernel<<<BATCH * HEADS, 352, LDS_ATTN, stream>>>(
      x, qwb, posb, agb, dwc_w, dwc_b, ybuf);
  proj_kernel<<<BATCH, 352, LDS_PROJ, stream>>>(ybuf, pwb, proj_b, outp);
}
